// TOGL_3152505996138
// MI455X (gfx1250) — compile-verified, matrix operand order checked by reading, not run on hardware
//
#include <hip/hip_runtime.h>
#include <hip/hip_bf16.h>

#define N_NODES   500000
#define N_EDGES   8000000
#define N_GRAPHS  512
#define N_FEAT    128
#define HIDDEN    128
#define N_FILT    8
#define OUT_DIM   64
#define BN_EPS    1e-5f

typedef __bf16 v16bf __attribute__((ext_vector_type(16)));
typedef __bf16 bf16x8 __attribute__((ext_vector_type(8)));
typedef float  v8f   __attribute__((ext_vector_type(8)));

union BF16Frag { v16bf v; bf16x8 h[2]; __bf16 b[16]; };

__device__ __forceinline__ __bf16 f2bf(float f) { return (__bf16)f; }

// A-fragment: 16x32 bf16 slice of row-major f32 matrix, per ISA layout.
// lane m (0..15) = row, half = K-half selector. p = &A[row*lda + kbase].
__device__ __forceinline__ BF16Frag load_afrag_f32(const float* __restrict__ p, int half) {
    BF16Frag a;
    const float* p0 = p + half * 8;
    #pragma unroll
    for (int i = 0; i < 8; ++i) { a.b[i] = f2bf(p0[i]); a.b[8 + i] = f2bf(p0[16 + i]); }
    return a;
}
// B-fragment from LDS bf16 row-major [n][k] with row stride ldb (bf16 units).
__device__ __forceinline__ BF16Frag load_bfrag_lds(const __bf16* __restrict__ bt, int half) {
    BF16Frag b;
    const __bf16* q = bt + half * 8;
    b.h[0] = *(const bf16x8*)q;
    b.h[1] = *(const bf16x8*)(q + 16);
    return b;
}

// ---------------------------------------------------------------------------
// Kernel 1: filtered_v = relu(x@W1+b1)@W2+b2   (WMMA bf16, f32 accumulate)
// Block = 256 threads = 8 waves; block covers 128 node rows; wave owns 16 rows.
// ---------------------------------------------------------------------------
#define LDS_STR 136

__global__ __launch_bounds__(256) void togl_k1_filtration(
    const float* __restrict__ x,  const float* __restrict__ W1,
    const float* __restrict__ b1, const float* __restrict__ W2,
    const float* __restrict__ b2, float* __restrict__ fv)
{
    __shared__ __bf16 smem0[128 * LDS_STR];   // w1t, later overlaid by hbuf
    __shared__ __bf16 w2t[16 * LDS_STR];

    const int tid = threadIdx.x;

    // W1t[n][k] = bf16(W1[k][n])
    for (int it = 0; it < 64; ++it) {
        int idx = it * 256 + tid;              // = k*128 + n
        int k = idx >> 7, n = idx & 127;
        smem0[n * LDS_STR + k] = f2bf(W1[idx]);
    }
    // W2t[n][k] = bf16(W2[k][n]) for n<8, else 0  (pad N to 16)
    for (int it = 0; it < 8; ++it) {
        int idx = it * 256 + tid;              // 16*128
        int n = idx >> 7, k = idx & 127;
        float v = (n < N_FILT) ? W2[k * N_FILT + n] : 0.0f;
        w2t[n * LDS_STR + k] = f2bf(v);
    }
    __syncthreads();

    const int lane = tid & 31;
    const int wave = tid >> 5;
    const int m    = lane & 15;
    const int half = lane >> 4;
    const int m0   = blockIdx.x * 128 + wave * 16;

    int arow = m0 + m;
    if (arow >= N_NODES) arow = N_NODES - 1;   // clamp; stores are guarded
    const float* xr = x + (size_t)arow * N_FEAT;

    v8f acc[8] = {};
    for (int kc = 0; kc < 4; ++kc) {
        BF16Frag a = load_afrag_f32(xr + kc * 32, half);
        #pragma unroll
        for (int nt = 0; nt < 8; ++nt) {
            BF16Frag b = load_bfrag_lds(&smem0[(nt * 16 + m) * LDS_STR + kc * 32], half);
            acc[nt] = __builtin_amdgcn_wmma_f32_16x16x32_bf16(
                false, a.v, false, b.v, (short)0, acc[nt], false, false);
        }
    }
    __syncthreads();   // all waves done reading w1t; reuse region as hbuf

    __bf16* hbuf = smem0;
    #pragma unroll
    for (int nt = 0; nt < 8; ++nt) {
        int col = nt * 16 + m;
        float bb = b1[col];
        #pragma unroll
        for (int r = 0; r < 8; ++r) {
            float v = acc[nt][r] + bb;
            v = v > 0.0f ? v : 0.0f;
            int row = wave * 16 + r + half * 8;
            hbuf[row * LDS_STR + col] = f2bf(v);
        }
    }
    __syncthreads();

    // GEMM2: h[16x128] @ W2t -> 16x16 tile (cols 0..7 valid)
    v8f acc2 = {};
    for (int kc = 0; kc < 4; ++kc) {
        BF16Frag a = load_bfrag_lds(&hbuf[(wave * 16 + m) * LDS_STR + kc * 32], half);
        BF16Frag b = load_bfrag_lds(&w2t[m * LDS_STR + kc * 32], half);
        acc2 = __builtin_amdgcn_wmma_f32_16x16x32_bf16(
            false, a.v, false, b.v, (short)0, acc2, false, false);
    }
    if (m < N_FILT) {
        float bb = b2[m];
        #pragma unroll
        for (int r = 0; r < 8; ++r) {
            int grow = m0 + r + half * 8;
            if (grow < N_NODES) fv[(size_t)grow * N_FILT + m] = acc2[r] + bb;
        }
    }
}

// ---------------------------------------------------------------------------
// Kernel 2: deaths gather + x0 assembly + y = relu(x0@W0+b0); segment sums.
// ---------------------------------------------------------------------------
__global__ __launch_bounds__(256) void togl_k2_x0(
    const float* __restrict__ fv, const int* __restrict__ eidx,
    const int* __restrict__ death, const int* __restrict__ batch,
    const float* __restrict__ W0, const float* __restrict__ b0,
    float* __restrict__ y, float* __restrict__ sums1, float* __restrict__ cnt)
{
    __shared__ float w0s[16 * 64];
    __shared__ float part[64];
    __shared__ int gmn, gmx, pcnt;
    const int tid = threadIdx.x;
    const int base = blockIdx.x * 256;
    for (int it = tid; it < 1024; it += 256) w0s[it] = W0[it];
    for (int it = tid; it < 64;   it += 256) part[it] = 0.0f;
    if (tid == 0) {
        int last = base + 255; if (last >= N_NODES) last = N_NODES - 1;
        gmn = batch[base]; gmx = batch[last]; pcnt = 0;   // batch is sorted
    }
    __syncthreads();

    const int i = base + tid;
    const bool active = (i < N_NODES);
    const bool uniform = (gmn == gmx);
    int g = 0;
    if (active) {
        g = batch[i];
        atomicAdd(&pcnt, 1);
        const int e = death[i];
        const int u = eidx[e];
        const int v = eidx[N_EDGES + e];
        float x0[16];
        #pragma unroll
        for (int f = 0; f < N_FILT; ++f) {
            x0[2 * f]     = fv[(size_t)i * N_FILT + f];
            x0[2 * f + 1] = fmaxf(fv[(size_t)u * N_FILT + f], fv[(size_t)v * N_FILT + f]);
        }
        for (int j = 0; j < 64; ++j) {
            float a = b0[j];
            #pragma unroll
            for (int k = 0; k < 16; ++k) a += x0[k] * w0s[k * 64 + j];
            a = fmaxf(a, 0.0f);
            y[(size_t)i * 64 + j] = a;
            if (uniform) atomicAdd(&part[j], a);
            else         atomicAdd(&sums1[(size_t)g * 64 + j], a);
        }
        if (!uniform) atomicAdd(&cnt[g], 1.0f);
    }
    __syncthreads();
    if (uniform) {
        if (tid < 64) atomicAdd(&sums1[(size_t)gmn * 64 + tid], part[tid]);
        if (tid == 0) atomicAdd(&cnt[gmn], (float)pcnt);
    }
}

// ---------------------------------------------------------------------------
// Kernel 3/5: xm = (sums / max(cnt,1)) @ LW
// ---------------------------------------------------------------------------
__global__ void togl_k3_xm1(const float* __restrict__ sums1, const float* __restrict__ cnt,
                            const float* __restrict__ l1W, float* __restrict__ xm1)
{
    __shared__ float mean[64];
    const int g = blockIdx.x, j = threadIdx.x;   // block = 64
    mean[j] = sums1[(size_t)g * 64 + j] / fmaxf(cnt[g], 1.0f);
    __syncthreads();
    float a = 0.0f;
    #pragma unroll
    for (int k = 0; k < 64; ++k) a += mean[k] * l1W[k * 64 + j];
    xm1[(size_t)g * 64 + j] = a;
}

__global__ void togl_k5_xm2(const float* __restrict__ sums2, const float* __restrict__ cnt,
                            const float* __restrict__ l2W, float* __restrict__ xm2)
{
    __shared__ float mean[64];
    const int g = blockIdx.x, j = threadIdx.x;   // block = 128
    if (j < 64) mean[j] = sums2[(size_t)g * 64 + j] / fmaxf(cnt[g], 1.0f);
    __syncthreads();
    float a = 0.0f;
    #pragma unroll
    for (int k = 0; k < 64; ++k) a += mean[k] * l2W[k * 128 + j];
    xm2[(size_t)g * 128 + j] = a;
}

// ---------------------------------------------------------------------------
// Kernel 4 (WMMA): x1 = relu(y@g1W + g1b - xm1[batch]) in-place; sums2.
// [N,64]x[64,64]; block covers 128 rows, wave 16 rows, 4 col tiles, K=64.
// ---------------------------------------------------------------------------
#define STR64 72

__global__ __launch_bounds__(256) void togl_k4_ds1(
    const int* __restrict__ batch, const float* __restrict__ g1W,
    const float* __restrict__ g1b, const float* __restrict__ xm1,
    float* __restrict__ y, float* __restrict__ sums2)
{
    __shared__ __bf16 bt[64 * STR64];
    __shared__ float part[64];
    __shared__ int gmn, gmx;
    const int tid  = threadIdx.x;
    const int base = blockIdx.x * 128;
    for (int it = tid; it < 4096; it += 256) {
        int k = it >> 6, n = it & 63;
        bt[n * STR64 + k] = f2bf(g1W[it]);
    }
    for (int it = tid; it < 64; it += 256) part[it] = 0.0f;
    if (tid == 0) {
        int last = base + 127; if (last >= N_NODES) last = N_NODES - 1;
        gmn = batch[base]; gmx = batch[last];    // batch is sorted
    }
    __syncthreads();

    const int lane = tid & 31, wave = tid >> 5;
    const int m = lane & 15, half = lane >> 4;
    const int m0 = base + wave * 16;
    int arow = m0 + m; if (arow >= N_NODES) arow = N_NODES - 1;
    const float* yr = y + (size_t)arow * 64;

    v8f acc[4] = {};
    for (int kc = 0; kc < 2; ++kc) {
        BF16Frag a = load_afrag_f32(yr + kc * 32, half);
        #pragma unroll
        for (int nt = 0; nt < 4; ++nt) {
            BF16Frag b = load_bfrag_lds(&bt[(nt * 16 + m) * STR64 + kc * 32], half);
            acc[nt] = __builtin_amdgcn_wmma_f32_16x16x32_bf16(
                false, a.v, false, b.v, (short)0, acc[nt], false, false);
        }
    }

    const bool uniform = (gmn == gmx);
    int gr[8];
    #pragma unroll
    for (int r = 0; r < 8; ++r) {
        int row = m0 + r + half * 8;
        gr[r] = (row < N_NODES) ? batch[row] : 0;
    }
    #pragma unroll
    for (int nt = 0; nt < 4; ++nt) {
        const int col = nt * 16 + m;
        const float gb = g1b[col];
        #pragma unroll
        for (int r = 0; r < 8; ++r) {
            int row = m0 + r + half * 8;
            if (row < N_NODES) {
                float v = acc[nt][r] + gb - xm1[(size_t)gr[r] * 64 + col];
                v = fmaxf(v, 0.0f);
                y[(size_t)row * 64 + col] = v;
                if (uniform) atomicAdd(&part[col], v);
                else         atomicAdd(&sums2[(size_t)gr[r] * 64 + col], v);
            }
        }
    }
    __syncthreads();
    if (uniform && tid < 64) atomicAdd(&sums2[(size_t)gmn * 64 + tid], part[tid]);
}

// ---------------------------------------------------------------------------
// Kernel 6 (WMMA): BN stats of h = relu(x1@g2W + g2b - xm2[batch]); h not stored.
// [N,64]x[64,128]; block 128 rows, wave 16 rows, 8 col tiles, K=64.
// ---------------------------------------------------------------------------
__global__ __launch_bounds__(256) void togl_k6_stats(
    const int* __restrict__ batch, const float* __restrict__ g2W,
    const float* __restrict__ g2b, const float* __restrict__ xm2,
    const float* __restrict__ x1, float* __restrict__ bn_sum, float* __restrict__ bn_sq)
{
    __shared__ __bf16 bt[128 * STR64];
    __shared__ float s1[128], s2[128];
    const int tid  = threadIdx.x;
    const int base = blockIdx.x * 128;
    for (int it = tid; it < 8192; it += 256) {
        int k = it >> 7, n = it & 127;
        bt[n * STR64 + k] = f2bf(g2W[it]);
    }
    for (int it = tid; it < 128; it += 256) { s1[it] = 0.0f; s2[it] = 0.0f; }
    __syncthreads();

    const int lane = tid & 31, wave = tid >> 5;
    const int m = lane & 15, half = lane >> 4;
    const int m0 = base + wave * 16;
    int arow = m0 + m; if (arow >= N_NODES) arow = N_NODES - 1;
    const float* xr = x1 + (size_t)arow * 64;

    v8f acc[8] = {};
    for (int kc = 0; kc < 2; ++kc) {
        BF16Frag a = load_afrag_f32(xr + kc * 32, half);
        #pragma unroll
        for (int nt = 0; nt < 8; ++nt) {
            BF16Frag b = load_bfrag_lds(&bt[(nt * 16 + m) * STR64 + kc * 32], half);
            acc[nt] = __builtin_amdgcn_wmma_f32_16x16x32_bf16(
                false, a.v, false, b.v, (short)0, acc[nt], false, false);
        }
    }
    int gr[8];
    #pragma unroll
    for (int r = 0; r < 8; ++r) {
        int row = m0 + r + half * 8;
        gr[r] = (row < N_NODES) ? batch[row] : 0;
    }
    #pragma unroll
    for (int nt = 0; nt < 8; ++nt) {
        const int col = nt * 16 + m;
        const float gb = g2b[col];
        #pragma unroll
        for (int r = 0; r < 8; ++r) {
            int row = m0 + r + half * 8;
            if (row < N_NODES) {
                float h = acc[nt][r] + gb - xm2[(size_t)gr[r] * 128 + col];
                h = fmaxf(h, 0.0f);
                atomicAdd(&s1[col], h);
                atomicAdd(&s2[col], h * h);
            }
        }
    }
    __syncthreads();
    if (tid < 128) { atomicAdd(&bn_sum[tid], s1[tid]); atomicAdd(&bn_sq[tid], s2[tid]); }
}

__global__ void togl_k7_bnfin(const float* __restrict__ bn_sum, const float* __restrict__ bn_sq,
                              float* __restrict__ mu, float* __restrict__ istd)
{
    const int j = threadIdx.x;   // 128
    const float inv_n = 1.0f / (float)N_NODES;
    float m = bn_sum[j] * inv_n;
    float v = bn_sq[j] * inv_n - m * m;
    mu[j]   = m;
    istd[j] = rsqrtf(fmaxf(v, 0.0f) + BN_EPS);
}

// ---------------------------------------------------------------------------
// Kernel 8 (WMMA): out = x + BN(relu(x1@g2W + g2b - xm2[batch]))
// ---------------------------------------------------------------------------
__global__ __launch_bounds__(256) void togl_k8_out(
    const int* __restrict__ batch, const float* __restrict__ g2W,
    const float* __restrict__ g2b, const float* __restrict__ xm2,
    const float* __restrict__ x1, const float* __restrict__ x,
    const float* __restrict__ gamma, const float* __restrict__ beta,
    const float* __restrict__ mu, const float* __restrict__ istd,
    float* __restrict__ out)
{
    __shared__ __bf16 bt[128 * STR64];
    const int tid  = threadIdx.x;
    const int base = blockIdx.x * 128;
    for (int it = tid; it < 8192; it += 256) {
        int k = it >> 7, n = it & 127;
        bt[n * STR64 + k] = f2bf(g2W[it]);
    }
    __syncthreads();

    const int lane = tid & 31, wave = tid >> 5;
    const int m = lane & 15, half = lane >> 4;
    const int m0 = base + wave * 16;
    int arow = m0 + m; if (arow >= N_NODES) arow = N_NODES - 1;
    const float* xr = x1 + (size_t)arow * 64;

    v8f acc[8] = {};
    for (int kc = 0; kc < 2; ++kc) {
        BF16Frag a = load_afrag_f32(xr + kc * 32, half);
        #pragma unroll
        for (int nt = 0; nt < 8; ++nt) {
            BF16Frag b = load_bfrag_lds(&bt[(nt * 16 + m) * STR64 + kc * 32], half);
            acc[nt] = __builtin_amdgcn_wmma_f32_16x16x32_bf16(
                false, a.v, false, b.v, (short)0, acc[nt], false, false);
        }
    }
    int gr[8];
    #pragma unroll
    for (int r = 0; r < 8; ++r) {
        int row = m0 + r + half * 8;
        gr[r] = (row < N_NODES) ? batch[row] : 0;
    }
    #pragma unroll
    for (int nt = 0; nt < 8; ++nt) {
        const int col = nt * 16 + m;
        const float gb  = g2b[col];
        const float muc = mu[col], isc = istd[col];
        const float gac = gamma[col], bec = beta[col];
        #pragma unroll
        for (int r = 0; r < 8; ++r) {
            int row = m0 + r + half * 8;
            if (row < N_NODES) {
                float h = acc[nt][r] + gb - xm2[(size_t)gr[r] * 128 + col];
                h = fmaxf(h, 0.0f);
                float hn = (h - muc) * isc;
                out[(size_t)row * 128 + col] = x[(size_t)row * 128 + col] + hn * gac + bec;
            }
        }
    }
}

// ---------------------------------------------------------------------------
extern "C" void kernel_launch(void* const* d_in, const int* in_sizes, int n_in,
                              void* d_out, int out_size, void* d_ws, size_t ws_size,
                              hipStream_t stream)
{
    const float* x     = (const float*)d_in[0];
    const float* W1    = (const float*)d_in[1];
    const float* b1    = (const float*)d_in[2];
    const float* W2    = (const float*)d_in[3];
    const float* b2    = (const float*)d_in[4];
    const float* W0    = (const float*)d_in[5];
    const float* b0    = (const float*)d_in[6];
    const float* g1W   = (const float*)d_in[7];
    const float* g1b   = (const float*)d_in[8];
    const float* l1W   = (const float*)d_in[9];
    const float* g2W   = (const float*)d_in[10];
    const float* g2b   = (const float*)d_in[11];
    const float* l2W   = (const float*)d_in[12];
    const float* gamma = (const float*)d_in[13];
    const float* beta  = (const float*)d_in[14];
    const int*   eidx  = (const int*)d_in[15];
    const int*   batch = (const int*)d_in[16];
    const int*   death = (const int*)d_in[17];
    float* out = (float*)d_out;

    size_t off = 0;
    auto alloc = [&](size_t bytes) -> void* {
        void* p = (char*)d_ws + off;
        off += (bytes + 255) & ~(size_t)255;
        return p;
    };
    float* fv     = (float*)alloc((size_t)N_NODES * N_FILT * 4);
    float* ybuf   = (float*)alloc((size_t)N_NODES * 64 * 4);      // y, then x1 in-place
    float* sums1  = (float*)alloc((size_t)N_GRAPHS * 64 * 4);
    float* sums2  = (float*)alloc((size_t)N_GRAPHS * 64 * 4);
    float* cnt    = (float*)alloc((size_t)N_GRAPHS * 4);
    float* xm1    = (float*)alloc((size_t)N_GRAPHS * 64 * 4);
    float* xm2    = (float*)alloc((size_t)N_GRAPHS * 128 * 4);
    float* bn_sum = (float*)alloc(128 * 4);
    float* bn_sq  = (float*)alloc(128 * 4);
    float* mu     = (float*)alloc(128 * 4);
    float* istd   = (float*)alloc(128 * 4);

    hipMemsetAsync(sums1,  0, (size_t)N_GRAPHS * 64 * 4, stream);
    hipMemsetAsync(sums2,  0, (size_t)N_GRAPHS * 64 * 4, stream);
    hipMemsetAsync(cnt,    0, (size_t)N_GRAPHS * 4, stream);
    hipMemsetAsync(bn_sum, 0, 128 * 4, stream);
    hipMemsetAsync(bn_sq,  0, 128 * 4, stream);

    const int gridNodes256 = (N_NODES + 255) / 256;
    const int gridNodes128 = (N_NODES + 127) / 128;

    togl_k1_filtration<<<gridNodes128, 256, 0, stream>>>(x, W1, b1, W2, b2, fv);
    togl_k2_x0<<<gridNodes256, 256, 0, stream>>>(fv, eidx, death, batch, W0, b0,
                                                 ybuf, sums1, cnt);
    togl_k3_xm1<<<N_GRAPHS, 64, 0, stream>>>(sums1, cnt, l1W, xm1);
    togl_k4_ds1<<<gridNodes128, 256, 0, stream>>>(batch, g1W, g1b, xm1, ybuf, sums2);
    togl_k5_xm2<<<N_GRAPHS, 128, 0, stream>>>(sums2, cnt, l2W, xm2);
    togl_k6_stats<<<gridNodes128, 256, 0, stream>>>(batch, g2W, g2b, xm2, ybuf,
                                                    bn_sum, bn_sq);
    togl_k7_bnfin<<<1, 128, 0, stream>>>(bn_sum, bn_sq, mu, istd);
    togl_k8_out<<<gridNodes128, 256, 0, stream>>>(batch, g2W, g2b, xm2, ybuf, x,
                                                  gamma, beta, mu, istd, out);
}
